// AttentionPooling_4114578669799
// MI455X (gfx1250) — matrix operand order repaired and checked by reading, not compiled, not run on hardware
//
#include <hip/hip_runtime.h>
#include <hip/hip_bf16.h>

typedef __attribute__((ext_vector_type(16))) _Float16 v16h;
typedef __attribute__((ext_vector_type(8)))  float    v8f;

#define N_BATCH 32
#define T_SEQ   4096
#define D_MODEL 256
#define N_HEADS 8
#define DH      32

// Dynamic-LDS layout (bytes):
//   [0,      131072)  Sexp : float[T_SEQ][N_HEADS]   scores -> probabilities
//   [131072, 139264)  Bs   : _Float16[16][256]       u (key-folded query), heads 8..15 zero
//   [139264, 140288)  q0   : float[256]
//   [140288, 141312)  v0   : float[256]
//   [141312, 142336)  red  : float[8][32]            reduction scratch
//   [142336, 142400)  cs   : float[8]                per-head bias term
//   [142400, 142464)  mh   : float[8]                per-head max
//   [142464, 142528)  ih   : float[8]                per-head 1/sum
#define SMEM_BYTES 142592

__global__ __launch_bounds__(256)
void attnpool_q0_kernel(const float* __restrict__ v,
                        const int*   __restrict__ mask,
                        const float* __restrict__ W,   // (256, 512) row-major
                        const float* __restrict__ b,   // (512)
                        float* __restrict__ out)       // (32, 256)
{
    extern __shared__ char smem[];
    float*    Sexp = (float*)smem;
    _Float16* Bs   = (_Float16*)(smem + 131072);
    float*    q0   = (float*)(smem + 139264);
    float*    v0   = (float*)(smem + 140288);
    float*    red  = (float*)(smem + 141312);
    float*    cs   = (float*)(smem + 142336);
    float*    mh   = (float*)(smem + 142400);
    float*    ih   = (float*)(smem + 142464);

    const int n   = blockIdx.x;
    const int tid = threadIdx.x;
    const float* vn = v + (size_t)n * T_SEQ * D_MODEL;

    // ---------------- Phase 0a: stage v[n,0,:] ----------------
    v0[tid] = vn[tid];
    __syncthreads();

    // ---------------- Phase 0b: q0 = v[n,0,:] @ W_q + b_q (coalesced over cols) ----
    {
        float acc = b[tid];
        #pragma unroll 4
        for (int i = 0; i < D_MODEL; ++i)
            acc = fmaf(v0[i], W[i * (2 * D_MODEL) + tid], acc);
        q0[tid] = acc;
    }
    __syncthreads();

    // ---------------- Phase 0c: u[h][i] -> f16 B tile in LDS; c[h] ----------------
    {
        const int i = tid;
        const float* Wr = W + (size_t)i * (2 * D_MODEL) + D_MODEL;  // key half, row i
        #pragma unroll
        for (int h = 0; h < N_HEADS; ++h) {
            float acc = 0.f;
            #pragma unroll 8
            for (int d = 0; d < DH; ++d)
                acc = fmaf(Wr[h * DH + d], q0[h * DH + d], acc);
            Bs[h * D_MODEL + i] = (_Float16)acc;
        }
        #pragma unroll
        for (int h = N_HEADS; h < 16; ++h)     // zero-pad unused WMMA columns
            Bs[h * D_MODEL + i] = (_Float16)0.f;
        if (tid < N_HEADS) {
            float acc = 0.f;
            #pragma unroll 8
            for (int d = 0; d < DH; ++d)
                acc = fmaf(b[D_MODEL + tid * DH + d], q0[tid * DH + d], acc);
            cs[tid] = acc;
        }
    }
    __syncthreads();

    // ---------------- Phase 1: scores via v_wmma_f32_16x16x32_f16 ----------------
    // Per wave: 16 t-rows x 16 head-cols (8 valid), K=256 in 8 chained WMMAs.
    // B (the key-folded query) is invariant across all t-tiles: hoist into VGPRs
    // once (8 x v16h = 64 VGPRs) so the hot loop is global-load -> cvt -> wmma only.
    {
        const int wave    = tid >> 5;
        const int lane    = tid & 31;
        const int mrow    = lane & 15;          // A row index / B column (head) index
        const int half_hi = lane >> 4;          // lane group 0 or 1
        const float cinit = (mrow < N_HEADS) ? cs[mrow] : 0.f;

        // B (32x16 f16 per step): lanes 0-15 hold K {0..15}, lanes 16-31 K {16..31}
        v16h breg[D_MODEL / 32];
        #pragma unroll
        for (int kc = 0; kc < D_MODEL / 32; ++kc)
            breg[kc] = *(const v16h*)(Bs + mrow * D_MODEL + kc * 32 + half_hi * 16);

        for (int it = 0; it < (T_SEQ / 16) / 8; ++it) {   // 32 tiles per wave
            const int t0 = (wave * 32 + it) * 16;
            const float* arow = vn + (size_t)(t0 + mrow) * D_MODEL;

            v8f acc;
            #pragma unroll
            for (int r = 0; r < 8; ++r) acc[r] = cinit;

            #pragma unroll
            for (int kc = 0; kc < D_MODEL / 32; ++kc) {
                // A (16-bit, 16x32): lanes 0-15 hold K {0..15}, lanes 16-31 K {16..31}
                const int k0 = kc * 32 + half_hi * 16;
                const float4 f0 = *(const float4*)(arow + k0 + 0);
                const float4 f1 = *(const float4*)(arow + k0 + 4);
                const float4 f2 = *(const float4*)(arow + k0 + 8);
                const float4 f3 = *(const float4*)(arow + k0 + 12);
                v16h a;
                a[0]  = (_Float16)f0.x; a[1]  = (_Float16)f0.y;
                a[2]  = (_Float16)f0.z; a[3]  = (_Float16)f0.w;
                a[4]  = (_Float16)f1.x; a[5]  = (_Float16)f1.y;
                a[6]  = (_Float16)f1.z; a[7]  = (_Float16)f1.w;
                a[8]  = (_Float16)f2.x; a[9]  = (_Float16)f2.y;
                a[10] = (_Float16)f2.z; a[11] = (_Float16)f2.w;
                a[12] = (_Float16)f3.x; a[13] = (_Float16)f3.y;
                a[14] = (_Float16)f3.z; a[15] = (_Float16)f3.w;

                acc = __builtin_amdgcn_wmma_f32_16x16x32_f16(
                        /*neg_a=*/false, a, /*neg_b=*/false, breg[kc],
                        /*c_mod=*/(short)0, acc, /*reuse_a=*/false, /*reuse_b=*/false);
            }
            if (mrow < N_HEADS) {               // D: VGPR r -> row r (+8 for hi lanes)
                #pragma unroll
                for (int r = 0; r < 8; ++r) {
                    const int trow = t0 + r + half_hi * 8;
                    Sexp[trow * N_HEADS + mrow] = acc[r] * 0.0625f;   // /sqrt(256)
                }
            }
        }
    }
    __syncthreads();

    // ---------------- Phase 2: softmax over T per head ----------------
    {
        const int h    = tid & 7;
        const int slot = tid >> 3;              // 0..31
        float m = -1e30f;
        for (int j = 0; j < T_SEQ / 32; ++j)
            m = fmaxf(m, Sexp[(slot + 32 * j) * N_HEADS + h]);
        red[h * 32 + slot] = m;
        __syncthreads();
        if (tid < N_HEADS) {
            float mm = -1e30f;
            #pragma unroll 8
            for (int s2 = 0; s2 < 32; ++s2) mm = fmaxf(mm, red[tid * 32 + s2]);
            mh[tid] = mm;
        }
        __syncthreads();
        const float mhv = mh[h];
        float sum = 0.f;
        for (int j = 0; j < T_SEQ / 32; ++j) {
            const int t = slot + 32 * j;
            const float p = __expf(Sexp[t * N_HEADS + h] - mhv);
            Sexp[t * N_HEADS + h] = p;
            sum += p;
        }
        red[h * 32 + slot] = sum;
        __syncthreads();
        if (tid < N_HEADS) {
            float ss = 0.f;
            #pragma unroll 8
            for (int s2 = 0; s2 < 32; ++s2) ss += red[tid * 32 + s2];
            ih[tid] = 1.0f / ss;
        }
        __syncthreads();
    }

    // ---------------- Phase 3: out[n, h*32+d] = inv[h] * sum_t p[t,h] * v[n,t,h*32+d]
    // LDS prob read is a wave-wide broadcast (h constant per wave); v read coalesced,
    // and the whole of v is L2-resident (134 MB < 192 MB) from the score pass.
    {
        const int h = tid >> 5;
        const float inv = ih[h];
        const float* vcol = vn + tid;
        float acc = 0.f;
        for (int tb = 0; tb < T_SEQ; tb += 16) {
            if (tb + 32 < T_SEQ)
                __builtin_prefetch(vcol + (size_t)(tb + 32) * D_MODEL, 0, 1);
            #pragma unroll
            for (int t = tb; t < tb + 16; ++t)
                acc = fmaf(Sexp[t * N_HEADS + h], vcol[(size_t)t * D_MODEL], acc);
        }
        float res = acc * inv;
        if (mask[(size_t)n * T_SEQ] == 0)       // masked query row -> softmax of -inf -> NaN
            res = __int_as_float(0x7FC00000);
        out[n * D_MODEL + tid] = res;
    }
}

extern "C" void kernel_launch(void* const* d_in, const int* in_sizes, int n_in,
                              void* d_out, int out_size, void* d_ws, size_t ws_size,
                              hipStream_t stream) {
    const float* v    = (const float*)d_in[0];
    const int*   mask = (const int*)d_in[1];
    const float* W    = (const float*)d_in[2];
    const float* b    = (const float*)d_in[3];
    float*       out  = (float*)d_out;
    (void)in_sizes; (void)n_in; (void)out_size; (void)d_ws; (void)ws_size;

    attnpool_q0_kernel<<<dim3(N_BATCH), dim3(256), SMEM_BYTES, stream>>>(v, mask, W, b, out);
}
